// NonSpikingMultiHeadAttention_33200097198666
// MI455X (gfx1250) — matrix operand order repaired and checked by reading, not compiled, not run on hardware
//
#include <hip/hip_runtime.h>

// ---------------------------------------------------------------------------
// Fused MHA block for MI455X (gfx1250, wave32, WMMA 16x16x32 f16 -> f32 acc)
//
//   1. f32 -> f16 conversion of x and the four weight matrices (into d_ws)
//   2. k_gemm_ln  x3 : Y = LN(x @ W^T + b)*g + be  for Q, K, V
//        A-panel (16 x 1024) staged into LDS by the Tensor Data Mover
//        (tensor_load_to_lds + s_wait_tensorcnt), B tiles double-buffered
//        in registers so WMMAs overlap the L2-resident weight streaming.
//   3. k_attn        : causal flash attention, online softmax, P exchanged
//        through per-wave LDS tiles, V loads hoisted over the softmax.
//   4. k_gemm_ln     : out = LN(attnout @ Wp^T + bp)*g_p + be_p -> fp32 d_out
// ---------------------------------------------------------------------------

typedef __attribute__((ext_vector_type(16))) _Float16 v16h;
typedef __attribute__((ext_vector_type(8)))  float    v8f;
typedef __attribute__((ext_vector_type(4)))  _Float16 v4h;
typedef unsigned int u32x4 __attribute__((ext_vector_type(4)));
typedef int          i32x8 __attribute__((ext_vector_type(8)));
typedef int          i32x4 __attribute__((ext_vector_type(4)));

#define B_  4
#define S_  2048
#define E_  1024
#define H_  16
#define HD_ 64
#define M_  (B_ * S_)   /* 8192 tokens */

// -------------------------- operand loaders --------------------------------

__device__ __forceinline__ v16h ld16h_pair(const _Float16* p, int second_off) {
  union { v16h v; uint4 q[2]; } r;
  r.q[0] = *(const uint4*)(p);
  r.q[1] = *(const uint4*)(p + second_off);
  return r.v;
}

// A operand, 16x32 f16 (ISA 7.12.2): lane m = lane&15, hi = lane>>4.
// Lane holds K in {hi*8 .. hi*8+7} and {16+hi*8 .. 16+hi*8+7}.
__device__ __forceinline__ v16h load_a(const _Float16* base, int ld,
                                       int row0, int k0, int lane) {
  const _Float16* p = base + (size_t)(row0 + (lane & 15)) * ld + k0 + ((lane >> 4) << 3);
  return ld16h_pair(p, 16);
}

// B operand, 32x16 f16, fed from B^T rows (n-major, k contiguous):
// lane n = lane&15; lane holds K in {hi*8 .. hi*8+7} and {16+hi*8 .. 16+hi*8+7}.
__device__ __forceinline__ v16h load_b(const _Float16* baseT, int ld,
                                       int col0, int k0, int lane) {
  const _Float16* p = baseT + (size_t)(col0 + (lane & 15)) * ld + k0 + ((lane >> 4) << 3);
  return ld16h_pair(p, 16);
}

__device__ __forceinline__ v8f wmma32(v16h a, v16h b, v8f c) {
  return __builtin_amdgcn_wmma_f32_16x16x32_f16(false, a, false, b, (short)0, c,
                                                false, false);
}

__device__ __forceinline__ float rmax16(float v) {
#pragma unroll
  for (int m = 1; m < 16; m <<= 1) v = fmaxf(v, __shfl_xor(v, m, 32));
  return v;
}
__device__ __forceinline__ float rsum16(float v) {
#pragma unroll
  for (int m = 1; m < 16; m <<= 1) v += __shfl_xor(v, m, 32);
  return v;
}

// ---------------- Tensor Data Mover: 2D tile -> LDS ------------------------
// Packs a D# per ISA 8.3/8.4 (group0: count/lds/global/type, group1:
// data_size=2B, tensor dims, tile dims, dim0 stride) and issues the 6-arg
// (clang-23 / therock) tensor_load_to_lds. Groups 2/3 (and the trailing
// group) zero => plain 2D tensor, no iterate/gather/multicast.
__device__ __forceinline__ void tdm_load_2d_f16(const _Float16* gsrc,
                                                unsigned lds_byte_off,
                                                unsigned dim0, unsigned dim1,
                                                unsigned tile0, unsigned tile1,
                                                unsigned stride0_elems) {
  unsigned long long ga = (unsigned long long)(uintptr_t)gsrc;
  u32x4 g0;
  g0[0] = 1u;                                            // count=1 (valid D#)
  g0[1] = lds_byte_off;                                  // lds_addr [63:32]
  g0[2] = (unsigned)(ga & 0xffffffffu);                  // global_addr lo
  g0[3] = (unsigned)((ga >> 32) & 0x01ffffffu)           // global_addr[56:32]
        | (2u << 30);                                    // type=2 ("image")
  i32x8 g1;
  g1[0] = (int)(1u << 16);                               // data_size=1 -> 2B
  g1[1] = (int)((dim0 & 0xffffu) << 16);                 // tensor_dim0 lo16
  g1[2] = (int)((dim0 >> 16) | ((dim1 & 0xffffu) << 16));// dim0 hi / dim1 lo
  g1[3] = (int)((dim1 >> 16) | (tile0 << 16));           // dim1 hi / tile_dim0
  g1[4] = (int)(tile1 & 0xffffu);                        // tile_dim1, tile_dim2=0
  g1[5] = (int)stride0_elems;                            // dim0_stride lo32
  g1[6] = 0;                                             // stride hi bits
  g1[7] = 0;
  i32x4 z4 = { 0, 0, 0, 0 };
  i32x8 z8 = { 0, 0, 0, 0, 0, 0, 0, 0 };
  __builtin_amdgcn_tensor_load_to_lds(g0, g1, z4, z4, z8, 0);
}

// -------------------------- f32 -> f16 convert -----------------------------

__global__ __launch_bounds__(256) void k_cvt(const float4* __restrict__ in,
                                             v4h* __restrict__ out, int n4) {
  int i = blockIdx.x * blockDim.x + threadIdx.x;
  int st = gridDim.x * blockDim.x;
  for (; i < n4; i += st) {
    float4 f = in[i];
    v4h h;
    h.x = (_Float16)f.x; h.y = (_Float16)f.y;
    h.z = (_Float16)f.z; h.w = (_Float16)f.w;
    out[i] = h;
  }
}

// ------------------- GEMM (+bias, +full-row LayerNorm) ---------------------
// Block: 256 threads = 8 waves; tile 16 rows x 1024 cols (full row => LN
// stats are block-local). A panel staged by TDM into LDS; B tiles double-
// buffered in registers (loads for k+32 issued before WMMAs for k).
// mode 0/1: Q/K -> f16 [B,H,S,HD]
// mode 2  : V   -> f16 [B,H,HD,S] + fp32 [B,H,S,HD] (d_out part 2)
// mode 3  : out -> fp32 [B,S,E]  (d_out part 1)

__global__ __launch_bounds__(256)
void k_gemm_ln(const _Float16* __restrict__ X, const _Float16* __restrict__ Wt,
               const float* __restrict__ bias, const float* __restrict__ g,
               const float* __restrict__ be,
               _Float16* __restrict__ out_h, _Float16* __restrict__ out_t,
               float* __restrict__ out_f, int mode) {
  __shared__ _Float16 sA[16 * E_];      // 32 KB A panel (TDM destination)
  __shared__ float red_s[8][16];
  __shared__ float red_q[8][16];

  const int wave = threadIdx.x >> 5;
  const int lane = threadIdx.x & 31;
  const int cl = lane & 15;
  const int hi = lane >> 4;
  const int row0 = blockIdx.x * 16;
  const int colbase = wave * 128;

  // One wave drives the Tensor Data Mover for the whole 16x1024 A panel.
  if (wave == 0) {
    tdm_load_2d_f16(X + (size_t)row0 * E_,
                    (unsigned)(uintptr_t)(void*)sA,
                    /*dim0=*/E_, /*dim1=*/16,
                    /*tile0=*/E_, /*tile1=*/16,
                    /*stride0=*/E_);
    __builtin_amdgcn_s_wait_tensorcnt(0);
  }
  __syncthreads();

  v8f acc[8] = {};

  // prime B double-buffer with k=0 tiles
  v16h bcur[8];
#pragma unroll
  for (int t = 0; t < 8; ++t)
    bcur[t] = load_b(Wt, E_, colbase + t * 16, 0, lane);

#pragma unroll 2
  for (int k0 = 0; k0 < E_ - 32; k0 += 32) {
    v16h a = load_a(sA, E_, 0, k0, lane);          // LDS (ds_load_b128)
    __builtin_prefetch(Wt + (size_t)(colbase + cl) * E_ + k0 + 256, 0, 1);
    v16h bn[8];
#pragma unroll
    for (int t = 0; t < 8; ++t)
      bn[t] = load_b(Wt, E_, colbase + t * 16, k0 + 32, lane);
#pragma unroll
    for (int t = 0; t < 8; ++t)
      acc[t] = wmma32(a, bcur[t], acc[t]);
#pragma unroll
    for (int t = 0; t < 8; ++t)
      bcur[t] = bn[t];
  }
  { // tail: k = E_-32
    v16h a = load_a(sA, E_, 0, E_ - 32, lane);
#pragma unroll
    for (int t = 0; t < 8; ++t)
      acc[t] = wmma32(a, bcur[t], acc[t]);
  }

  // bias (per output column)
#pragma unroll
  for (int t = 0; t < 8; ++t) {
    float bv = bias[colbase + t * 16 + cl];
#pragma unroll
    for (int i = 0; i < 8; ++i) acc[t][i] += bv;
  }

  // per-row partial stats over this wave's 128 columns -> LDS
#pragma unroll
  for (int i = 0; i < 8; ++i) {
    float s = 0.f, s2 = 0.f;
#pragma unroll
    for (int t = 0; t < 8; ++t) { float v = acc[t][i]; s += v; s2 += v * v; }
    s = rsum16(s);
    s2 = rsum16(s2);
    if (cl == 0) { red_s[wave][i + 8 * hi] = s; red_q[wave][i + 8 * hi] = s2; }
  }
  __syncthreads();

  float mu[8], rs[8];
#pragma unroll
  for (int i = 0; i < 8; ++i) {
    int r = i + 8 * hi;
    float s = 0.f, s2 = 0.f;
#pragma unroll
    for (int w = 0; w < 8; ++w) { s += red_s[w][r]; s2 += red_q[w][r]; }
    float m = s * (1.0f / E_);
    float var = s2 * (1.0f / E_) - m * m;      // biased var, matches jnp.var
    mu[i] = m;
    rs[i] = rsqrtf(var + 1e-5f);
  }

#pragma unroll
  for (int t = 0; t < 8; ++t) {
    int col = colbase + t * 16 + cl;
    float gv = g[col], bev = be[col];
    int hh = col >> 6, dd = col & 63;
#pragma unroll
    for (int i = 0; i < 8; ++i) {
      float y = (acc[t][i] - mu[i]) * rs[i] * gv + bev;
      int r = row0 + i + 8 * hi;               // token index
      int bb = r >> 11, ss = r & (S_ - 1);
      if (mode <= 1) {
        out_h[((size_t)(bb * H_ + hh) * S_ + ss) * HD_ + dd] = (_Float16)y;
      } else if (mode == 2) {
        out_t[((size_t)(bb * H_ + hh) * HD_ + dd) * S_ + ss] = (_Float16)y;
        out_f[((size_t)(bb * H_ + hh) * S_ + ss) * HD_ + dd] = y;
      } else {
        out_f[(size_t)r * E_ + col] = y;
      }
    }
  }
}

// ----------------------- causal flash attention ----------------------------
// grid = (S/128, B*H); 8 waves each own a 16-row Q strip. Key tiles of 32.
// V B-operand loads are issued right after the score WMMAs so they are in
// flight across the softmax VALU work and the LDS P-exchange.

__global__ __launch_bounds__(256)
void k_attn(const _Float16* __restrict__ Qh, const _Float16* __restrict__ Kh,
            const _Float16* __restrict__ Vt, _Float16* __restrict__ AO) {
  __shared__ _Float16 Pl[8][16 * 32];   // per-wave P tile, 1KB each

  const int wave = threadIdx.x >> 5;
  const int lane = threadIdx.x & 31;
  const int cl = lane & 15;
  const int hi = lane >> 4;
  const int bh = blockIdx.y;
  const int b = bh >> 4;
  const int h = bh & (H_ - 1);

  const _Float16* Q = Qh + (size_t)bh * S_ * HD_;
  const _Float16* K = Kh + (size_t)bh * S_ * HD_;
  const _Float16* V = Vt + (size_t)bh * HD_ * S_;

  const int q0 = blockIdx.x * 128 + wave * 16;

  v16h aq0 = load_a(Q, HD_, q0, 0, lane);
  v16h aq1 = load_a(Q, HD_, q0, 32, lane);

  v8f o[4] = {};
  float mrow[8], lrow[8];
#pragma unroll
  for (int i = 0; i < 8; ++i) { mrow[i] = -1e30f; lrow[i] = 0.f; }

  const int kend = blockIdx.x * 128 + 128;   // causal bound for this q-tile
  const float scale = 0.125f;                // 1/sqrt(64)

  for (int kt = 0; kt < kend; kt += 32) {
    v8f s0 = {}, s1 = {};
    {
      v16h b0 = load_b(K, HD_, kt, 0, lane);
      v16h b1 = load_b(K, HD_, kt, 32, lane);
      s0 = wmma32(aq0, b0, s0);
      s0 = wmma32(aq1, b1, s0);
      v16h b2 = load_b(K, HD_, kt + 16, 0, lane);
      v16h b3 = load_b(K, HD_, kt + 16, 32, lane);
      s1 = wmma32(aq0, b2, s1);
      s1 = wmma32(aq1, b3, s1);
    }

    // V tiles for this key block: issue now, consume after softmax/LDS.
    v16h bv[4];
#pragma unroll
    for (int dt = 0; dt < 4; ++dt)
      bv[dt] = load_b(V, S_, dt * 16, kt, lane);

#pragma unroll
    for (int i = 0; i < 8; ++i) {
      int qr = q0 + i + 8 * hi;
      float v0 = (kt + cl      <= qr) ? s0[i] * scale : -1e30f;
      float v1 = (kt + 16 + cl <= qr) ? s1[i] * scale : -1e30f;
      float tm = rmax16(fmaxf(v0, v1));
      float mn = fmaxf(mrow[i], tm);
      float corr = __expf(mrow[i] - mn);
      mrow[i] = mn;
      float p0 = __expf(v0 - mn);
      float p1 = __expf(v1 - mn);
      lrow[i] = lrow[i] * corr + rsum16(p0 + p1);
      o[0][i] *= corr; o[1][i] *= corr; o[2][i] *= corr; o[3][i] *= corr;
      s0[i] = p0;
      s1[i] = p1;
    }

    // C-layout f32 -> A-layout f16 via per-wave LDS tile (row-major 16x32)
    _Float16* pw = &Pl[wave][0];
#pragma unroll
    for (int i = 0; i < 8; ++i) {
      int r = i + 8 * hi;
      pw[r * 32 + cl]      = (_Float16)s0[i];
      pw[r * 32 + 16 + cl] = (_Float16)s1[i];
    }
    asm volatile("s_wait_dscnt 0" ::: "memory");  // DS is in-order per wave
    v16h pa = load_a(pw, 32, 0, 0, lane);

#pragma unroll
    for (int dt = 0; dt < 4; ++dt)
      o[dt] = wmma32(pa, bv[dt], o[dt]);
  }

  // normalize and store attn-out in [B,S,E] layout (f16, feeds P projection)
#pragma unroll
  for (int i = 0; i < 8; ++i) {
    float inv = 1.0f / lrow[i];
    int qr = q0 + i + 8 * hi;
    size_t trow = (size_t)(b * S_ + qr) * E_;
#pragma unroll
    for (int dt = 0; dt < 4; ++dt) {
      AO[trow + h * HD_ + dt * 16 + cl] = (_Float16)(o[dt][i] * inv);
    }
  }
}

// ------------------------------ launcher -----------------------------------

extern "C" void kernel_launch(void* const* d_in, const int* in_sizes, int n_in,
                              void* d_out, int out_size, void* d_ws, size_t ws_size,
                              hipStream_t stream) {
  (void)in_sizes; (void)n_in; (void)out_size; (void)ws_size;

  const float* x    = (const float*)d_in[0];
  const float* wq   = (const float*)d_in[1];
  const float* bq   = (const float*)d_in[2];
  const float* wk   = (const float*)d_in[3];
  const float* bk   = (const float*)d_in[4];
  const float* wv   = (const float*)d_in[5];
  const float* bv   = (const float*)d_in[6];
  const float* wp   = (const float*)d_in[7];
  const float* bp   = (const float*)d_in[8];
  const float* g_q  = (const float*)d_in[9];
  const float* be_q = (const float*)d_in[10];
  const float* g_k  = (const float*)d_in[11];
  const float* be_k = (const float*)d_in[12];
  const float* g_v  = (const float*)d_in[13];
  const float* be_v = (const float*)d_in[14];
  const float* g_p  = (const float*)d_in[15];
  const float* be_p = (const float*)d_in[16];

  // workspace layout (f16 elements); AO aliases xh (x dead after V GEMM)
  _Float16* ws = (_Float16*)d_ws;
  size_t off = 0;
  _Float16* xh  = ws + off; off += (size_t)M_ * E_;   // 16 MB, reused as AO
  _Float16* wqh = ws + off; off += (size_t)E_ * E_;
  _Float16* wkh = ws + off; off += (size_t)E_ * E_;
  _Float16* wvh = ws + off; off += (size_t)E_ * E_;
  _Float16* wph = ws + off; off += (size_t)E_ * E_;
  _Float16* Qh  = ws + off; off += (size_t)M_ * E_;
  _Float16* Kh  = ws + off; off += (size_t)M_ * E_;
  _Float16* Vt  = ws + off; off += (size_t)M_ * E_;   // total 72 MB

  float* outP = (float*)d_out;                        // [B,S,E]
  float* outV = (float*)d_out + (size_t)M_ * E_;      // [B,H,S,HD]

  // 1) precision conversion
  k_cvt<<<1024, 256, 0, stream>>>((const float4*)x,  (v4h*)xh,  (M_ * E_) / 4);
  k_cvt<<<256,  256, 0, stream>>>((const float4*)wq, (v4h*)wqh, (E_ * E_) / 4);
  k_cvt<<<256,  256, 0, stream>>>((const float4*)wk, (v4h*)wkh, (E_ * E_) / 4);
  k_cvt<<<256,  256, 0, stream>>>((const float4*)wv, (v4h*)wvh, (E_ * E_) / 4);
  k_cvt<<<256,  256, 0, stream>>>((const float4*)wp, (v4h*)wph, (E_ * E_) / 4);

  // 2) fused projection + bias + layernorm (Q, K, V)
  k_gemm_ln<<<M_ / 16, 256, 0, stream>>>(xh, wqh, bq, g_q, be_q, Qh, nullptr, nullptr, 0);
  k_gemm_ln<<<M_ / 16, 256, 0, stream>>>(xh, wkh, bk, g_k, be_k, Kh, nullptr, nullptr, 1);
  k_gemm_ln<<<M_ / 16, 256, 0, stream>>>(xh, wvh, bv, g_v, be_v, nullptr, Vt, outV, 2);

  // 3) causal flash attention (writes f16 attn-out over xh)
  k_attn<<<dim3(S_ / 128, B_ * H_), 256, 0, stream>>>(Qh, Kh, Vt, xh);

  // 4) output projection + layernorm -> fp32 d_out
  k_gemm_ln<<<M_ / 16, 256, 0, stream>>>(xh, wph, bp, g_p, be_p, nullptr, nullptr, outP, 3);
}
